// SelfAttention_8804682957224
// MI455X (gfx1250) — matrix operand order repaired and checked by reading, not compiled, not run on hardware
//
#include <hip/hip_runtime.h>
#include <hip/hip_bf16.h>
#include <math.h>

// ---------------------------------------------------------------------------
// CDNA5 (gfx1250) self-attention forward, MI455X-tuned:
//  * all GEMMs through v_wmma_f32_16x16x32_bf16 (bf16 in, f32 accumulate)
//  * LDS-blocked GEMM (128x64 tile, 8 waves) with TDM (tensor_load_to_lds)
//    double-buffered staging where available
//  * attention: 16 waves = 16 heads per query block; head-independent
//    mask/shift tiles staged in LDS ONCE per key block (16x traffic cut)
// ---------------------------------------------------------------------------

typedef __attribute__((ext_vector_type(16))) __bf16       v16bf;
typedef __attribute__((ext_vector_type(8)))  float        v8f;
typedef __attribute__((ext_vector_type(4)))  unsigned int v4u;
typedef __attribute__((ext_vector_type(8)))  int          v8i;
typedef __attribute__((ext_vector_type(4)))  int          v4i;

#define WMMA_BF16(a, b, c) \
  __builtin_amdgcn_wmma_f32_16x16x32_bf16(false, (a), false, (b), (short)0, (c), false, false)

// --- Tensor Data Mover availability / arity (see CDNA5_HIP.md) -------------
#if defined(__has_builtin)
#if __has_builtin(__builtin_amdgcn_tensor_load_to_lds)
#define USE_TDM 1
#endif
#endif
#ifndef USE_TDM
#define USE_TDM 0
#endif
#if __has_include(<hip/amd_detail/amd_gfx1250_TDM.h>)
#define TDM_6ARG 1   // therock-10.0 headers -> clang-23 6-arg builtin
#endif

#if USE_TDM
// Issue a 2D TENSOR_LOAD_TO_LDS. D# layout per CDNA5 ISA
// §8.3-8.6. pad_enable inserts `pad_dwords` after every `row_dwords` DWORDs,
// producing the padded LDS row stride used by the compute side.
// row_dwords must be a power of two; pad encoding: interval v -> 2^(v+1).
__device__ __forceinline__ void tdm_load_2d(unsigned lds_byte_addr,
                                            const float* gaddr,
                                            unsigned tensor_d0, unsigned tensor_d1,
                                            unsigned long long stride0_elems,
                                            unsigned tile_d0, unsigned tile_d1,
                                            unsigned pad_interval_code,
                                            unsigned pad_amount_code) {
  const unsigned long long ga = (unsigned long long)(size_t)gaddr;
  v4u g0;
  g0[0] = 1u;                                           // count=1, user mode
  g0[1] = lds_byte_addr;                                // lds_addr
  g0[2] = (unsigned)(ga & 0xFFFFFFFFu);                 // global_addr[31:0]
  g0[3] = (unsigned)((ga >> 32) & 0x01FFFFFFu) | (2u << 30); // [56:32] | type=2
  v8i g1;
  g1[0] = (int)((2u << 16) | (1u << 20) |               // data_size=4B, pad_en
                (pad_interval_code << 22) | (pad_amount_code << 25));
  g1[1] = (int)((tensor_d0 & 0xFFFFu) << 16);           // tensor_dim0[15:0]
  g1[2] = (int)((tensor_d0 >> 16) | ((tensor_d1 & 0xFFFFu) << 16));
  g1[3] = (int)((tensor_d1 >> 16) | (tile_d0 << 16));   // | tile_dim0
  g1[4] = (int)(tile_d1 & 0xFFFFu);                     // tile_dim1 (dim2=0)
  g1[5] = (int)(stride0_elems & 0xFFFFFFFFull);         // dim0_stride[31:0]
  g1[6] = (int)((stride0_elems >> 32) & 0xFFFFull);     // [47:32] | s1_lo=0
  g1[7] = 0;
  v4i z4 = {};
#if defined(TDM_6ARG)
  v8i z8 = {};
  __builtin_amdgcn_tensor_load_to_lds(g0, g1, z4, z4, z8, 0);
#else
  __builtin_amdgcn_tensor_load_to_lds(g0, g1, z4, z4, 0);
#endif
}
#endif  // USE_TDM

// ---------------------------------------------------------------------------
// Blocked GEMM: C[M,N] = A[M,K] @ B[K,N] + bias[N]    (fp32 in memory)
// Workgroup: 256 threads (8 waves), tile 128(M) x 64(N), K-step 32.
// Waves in a 4x2 grid; each wave owns a 32x32 sub-tile = 4 WMMA accumulators.
// A/B K-panels staged in LDS (fp32, padded rows), double buffered via TDM.
//
// WMMA lane layouts used here (A and B share k order), half = lane>>4:
//   A 16x32: lane holds row lane&15; elems 0..7 = K 16*half+0..7,
//            elems 8..15 = K 16*half+8..15
//   B 32x16: lane holds col lane&15; elems j = K 16*half+j
//   C 16x16: lane holds col lane&15; reg r = row r+8*half
// ---------------------------------------------------------------------------
#define BM 128
#define BN 64
#define BK 32
#define A_STR (BK + 4)   // padded LDS row stride (f32) -> TDM pad 4 DWORDs
#define B_STR (BN + 4)

__global__ __launch_bounds__(256)
void gemm_bf16_wmma(const float* __restrict__ A, const float* __restrict__ B,
                    const float* __restrict__ bias, float* __restrict__ C,
                    int M, int N, int K) {
  __shared__ float As[2][BM * A_STR];   // 2*128*36*4 = 36864 B
  __shared__ float Bs[2][BK * B_STR];   // 2* 32*68*4 = 17408 B

  const int tid  = threadIdx.x;
  const int wave = tid >> 5;
  const int lane = tid & 31;
  const int half = lane >> 4;
  const int l15  = lane & 15;
  const int bm0  = blockIdx.y * BM;
  const int bn0  = blockIdx.x * BN;
  const int wm   = (wave >> 1) * 32;    // wave M offset in block tile
  const int wn   = (wave & 1) * 32;     // wave N offset in block tile

  v8f acc[2][2] = {};
  const int nK = K / BK;

  // ---- stage K-panel `ik` into buffer ik&1 ----
  auto stage = [&](int ik) {
    const int k0 = ik * BK;
#if USE_TDM
    if (tid < 32) {   // one wave issues both DMAs (EXEC ignored by TDM)
      tdm_load_2d((unsigned)(size_t)&As[ik & 1][0],
                  A + (size_t)bm0 * K + k0,
                  (unsigned)K, (unsigned)M, (unsigned long long)K,
                  BK, BM, /*interval 2^5=32*/ 4u, /*pad 4 DW*/ 3u);
      tdm_load_2d((unsigned)(size_t)&Bs[ik & 1][0],
                  B + (size_t)k0 * N + bn0,
                  (unsigned)N, (unsigned)K, (unsigned long long)N,
                  BN, BK, /*interval 2^6=64*/ 5u, /*pad 4 DW*/ 3u);
    }
#else
    {  // cooperative fallback copy, honoring the padded LDS strides
      const int ar = tid >> 1, ac = (tid & 1) * 16;           // A: 128x32
      const float* ag = A + (size_t)(bm0 + ar) * K + k0 + ac;
      float* al = &As[ik & 1][ar * A_STR + ac];
#pragma unroll
      for (int j = 0; j < 16; ++j) al[j] = ag[j];
      const int br = tid >> 3, bc = (tid & 7) * 8;            // B: 32x64
      const float* bg = B + (size_t)(k0 + br) * N + bn0 + bc;
      float* bl = &Bs[ik & 1][br * B_STR + bc];
#pragma unroll
      for (int j = 0; j < 8; ++j) bl[j] = bg[j];
    }
#endif
  };

  stage(0);
  for (int ik = 0; ik < nK; ++ik) {
    if (ik + 1 < nK) {
      stage(ik + 1);                       // prefetch next panel (other buf)
#if USE_TDM
      if (tid < 32) __builtin_amdgcn_s_wait_tensorcnt(2);  // this panel done
#endif
    } else {
#if USE_TDM
      if (tid < 32) __builtin_amdgcn_s_wait_tensorcnt(0);
#endif
    }
    __syncthreads();

    const float* as = &As[ik & 1][0];
    const float* bs = &Bs[ik & 1][0];

    v16bf afrag[2];
#pragma unroll
    for (int mt = 0; mt < 2; ++mt) {
      const float* ap = as + (wm + mt * 16 + l15) * A_STR + 16 * half;
#pragma unroll
      for (int j = 0; j < 8; ++j) {
        afrag[mt][j]     = (__bf16)ap[j];
        afrag[mt][8 + j] = (__bf16)ap[8 + j];
      }
    }
    v16bf bfrag[2];
#pragma unroll
    for (int nt = 0; nt < 2; ++nt) {
      const float* bp = bs + (16 * half) * B_STR + wn + nt * 16 + l15;
#pragma unroll
      for (int j = 0; j < 16; ++j) bfrag[nt][j] = (__bf16)bp[j * B_STR];
    }
#pragma unroll
    for (int mt = 0; mt < 2; ++mt)
#pragma unroll
      for (int nt = 0; nt < 2; ++nt)
        acc[mt][nt] = WMMA_BF16(afrag[mt], bfrag[nt], acc[mt][nt]);

    __syncthreads();
  }

#pragma unroll
  for (int mt = 0; mt < 2; ++mt)
#pragma unroll
    for (int nt = 0; nt < 2; ++nt) {
      const int col = bn0 + wn + nt * 16 + l15;
      const float bv = bias ? bias[col] : 0.0f;
#pragma unroll
      for (int r = 0; r < 8; ++r) {
        const int row = bm0 + wm + mt * 16 + r + 8 * half;
        C[(size_t)row * N + col] = acc[mt][nt][r] + bv;
      }
    }
}

// ---------------------------------------------------------------------------
// Scatter k_out / v_out from the packed qkv buffer.
// ---------------------------------------------------------------------------
__global__ __launch_bounds__(256)
void split_kv(const float* __restrict__ qkv, float* __restrict__ k_out,
              float* __restrict__ v_out) {
  const int idx = blockIdx.x * 256 + threadIdx.x;   // over T*D
  const int t = idx >> 10;
  const int c = idx & 1023;
  const int h = c >> 6;
  const int d = c & 63;
  const float* row = qkv + (size_t)t * 3072 + h * 192;
  k_out[idx] = row[64 + d];
  v_out[idx] = row[128 + d];
}

// ---------------------------------------------------------------------------
// Attention: one workgroup (512 threads = 16 waves) per 16-query block;
// wave w handles head w. mask/shift are head-independent -> staged in LDS
// once per 32-key block and shared by all 16 heads. All waves share the same
// causal trip count (same query block) so barriers are uniform.
// ---------------------------------------------------------------------------
__global__ __launch_bounds__(512)
void attn_head(const float* __restrict__ qkv, const float* __restrict__ mask,
               const float* __restrict__ shift, float* __restrict__ att_out) {
  const int T = 2048;
  const int qb   = blockIdx.x;        // 0..127
  const int tid  = threadIdx.x;
  const int h    = tid >> 5;          // wave index == head
  const int lane = tid & 31;
  const int half = lane >> 4;
  const int l15  = lane & 15;
  const int ka   = 16 * half;

  const float inv_sqrt = 0.125f;                   // 1/sqrt(64)
  const float scale_h  = exp2f(-0.5f * (float)h);  // exp(-ln2 * h/2)

  __shared__ float   mask_s [16][32];
  __shared__ float   shift_s[16][32];
  __shared__ __bf16  pshare[16][16][32];           // per-wave P staging

  // ---- Q fragment (A layout), dh=64 as two K=32 chunks ----
  const int qrow = qb * 16 + l15;
  const float* qp = qkv + (size_t)qrow * 3072 + h * 192;
  v16bf qa0, qa1;
#pragma unroll
  for (int j = 0; j < 8; ++j) {
    qa0[j]     = (__bf16)qp[ka + j];
    qa0[8 + j] = (__bf16)qp[ka + 8 + j];
    qa1[j]     = (__bf16)qp[32 + ka + j];
    qa1[8 + j] = (__bf16)qp[32 + ka + 8 + j];
  }

  const int crow_base = qb * 16 + 8 * half;   // C-layout row = crow_base + r
  float mrun[8], lrun[8];
  v8f o[4] = {};
#pragma unroll
  for (int r = 0; r < 8; ++r) { mrun[r] = -3.0e38f; lrun[r] = 0.0f; }

  const int srow = tid >> 5 & 15;   // staging row (== h, 0..15)
  const int kb_max = (qb * 16 + 15) / 32;     // causal skip (head-independent)
  for (int kb = 0; kb <= kb_max; ++kb) {
    const int kbase = kb * 32;

    // ---- stage mask/shift 16x32 tiles ONCE for all 16 heads ----
    __syncthreads();   // previous iteration's readers are done
    {
      const int col = lane;  // wave w stages row w, cols 0..31
      mask_s [srow][col] = mask [(size_t)(qb * 16 + srow) * T + kbase + col];
      shift_s[srow][col] = shift[(size_t)(qb * 16 + srow) * T + kbase + col];
    }
    __syncthreads();

    // ---- S tiles: keys [kbase,kbase+16) and [kbase+16,kbase+32) ----
    v8f s0 = {}, s1 = {};
    const float* kp0 = qkv + (size_t)(kbase + l15) * 3072 + h * 192 + 64;
    const float* kp1 = kp0 + (size_t)16 * 3072;
    {
      v16bf b;
#pragma unroll
      for (int j = 0; j < 16; ++j) b[j] = (__bf16)kp0[16 * half + j];
      s0 = WMMA_BF16(qa0, b, s0);
#pragma unroll
      for (int j = 0; j < 16; ++j) b[j] = (__bf16)kp0[32 + 16 * half + j];
      s0 = WMMA_BF16(qa1, b, s0);
#pragma unroll
      for (int j = 0; j < 16; ++j) b[j] = (__bf16)kp1[16 * half + j];
      s1 = WMMA_BF16(qa0, b, s1);
#pragma unroll
      for (int j = 0; j < 16; ++j) b[j] = (__bf16)kp1[32 + 16 * half + j];
      s1 = WMMA_BF16(qa1, b, s1);
    }

    // ---- mask / shift from LDS (exact reference semantics, fp32) ----
#pragma unroll
    for (int r = 0; r < 8; ++r) {
      const int rr = r + 8 * half;
      const float m0  = mask_s [rr][l15];
      const float m1  = mask_s [rr][16 + l15];
      const float sh0 = shift_s[rr][l15];
      const float sh1 = shift_s[rr][16 + l15];
      s0[r] = s0[r] * inv_sqrt * m0 - 1.0e6f * (1.0f - m0) - scale_h * sh0;
      s1[r] = s1[r] * inv_sqrt * m1 - 1.0e6f * (1.0f - m1) - scale_h * sh1;
    }

    // ---- online softmax: each row lives in one half-wave at reg r ----
    float alpha[8];
#pragma unroll
    for (int r = 0; r < 8; ++r) {
      float vmax = fmaxf(s0[r], s1[r]);
#pragma unroll
      for (int m = 1; m < 16; m <<= 1) vmax = fmaxf(vmax, __shfl_xor(vmax, m, 32));
      const float mnew = fmaxf(mrun[r], vmax);
      alpha[r] = __expf(mrun[r] - mnew);
      mrun[r]  = mnew;
      const float p0 = __expf(s0[r] - mnew);
      const float p1 = __expf(s1[r] - mnew);
      s0[r] = p0; s1[r] = p1;
      float rs = p0 + p1;
#pragma unroll
      for (int m = 1; m < 16; m <<= 1) rs += __shfl_xor(rs, m, 32);
      lrun[r] = lrun[r] * alpha[r] + rs;
    }

    // ---- P: C layout -> per-wave LDS tile -> A layout ----
#pragma unroll
    for (int r = 0; r < 8; ++r) {
      pshare[h][r + 8 * half][l15]      = (__bf16)s0[r];
      pshare[h][r + 8 * half][16 + l15] = (__bf16)s1[r];
    }
    __syncthreads();   // uniform across waves (same trip count)
    v16bf pa;
#pragma unroll
    for (int j = 0; j < 8; ++j) {
      pa[j]     = pshare[h][l15][ka + j];
      pa[8 + j] = pshare[h][l15][ka + 8 + j];
    }

    // ---- O(16x64) update: rescale by alpha, then O += P @ V (K=32) ----
#pragma unroll
    for (int nb = 0; nb < 4; ++nb) {
#pragma unroll
      for (int r = 0; r < 8; ++r) o[nb][r] *= alpha[r];
      const float* vp = qkv + (size_t)(kbase + 16 * half) * 3072
                        + h * 192 + 128 + nb * 16 + l15;
      v16bf vb;
#pragma unroll
      for (int j = 0; j < 16; ++j) vb[j] = (__bf16)vp[(size_t)j * 3072];
      o[nb] = WMMA_BF16(pa, vb, o[nb]);
    }
  }

  // ---- finalize: divide by row sums, head-concatenated fp32 store ----
#pragma unroll
  for (int nb = 0; nb < 4; ++nb) {
#pragma unroll
    for (int r = 0; r < 8; ++r) {
      const int row = crow_base + r;
      att_out[(size_t)row * 1024 + h * 64 + nb * 16 + l15] = o[nb][r] / lrun[r];
    }
  }
}

// ---------------------------------------------------------------------------
// Launch. Inputs: x, mask, shift, W, b, Wo, bo (fp32).
// d_out = [o | k_out | v_out], each 2048*1024 fp32.
// Workspace: qkv [2048,3072] + att [2048,1024] fp32 (~33.5 MB).
// ---------------------------------------------------------------------------
extern "C" void kernel_launch(void* const* d_in, const int* in_sizes, int n_in,
                              void* d_out, int out_size, void* d_ws, size_t ws_size,
                              hipStream_t stream) {
  (void)in_sizes; (void)n_in; (void)out_size; (void)ws_size;
  const float* x     = (const float*)d_in[0];
  const float* mask  = (const float*)d_in[1];
  const float* shift = (const float*)d_in[2];
  const float* W     = (const float*)d_in[3];
  const float* b     = (const float*)d_in[4];
  const float* Wo    = (const float*)d_in[5];
  const float* bo    = (const float*)d_in[6];

  float* out_o = (float*)d_out;
  float* k_out = out_o + (size_t)2048 * 1024;
  float* v_out = k_out + (size_t)2048 * 1024;

  float* qkv = (float*)d_ws;                    // [2048, 3072]
  float* att = qkv + (size_t)2048 * 3072;       // [2048, 1024]

  // 1) qkv = x @ W + b    (M=2048, N=3072, K=1024)
  gemm_bf16_wmma<<<dim3(3072 / BN, 2048 / BM), 256, 0, stream>>>(
      x, W, b, qkv, 2048, 3072, 1024);

  // 2) scatter k_out / v_out
  split_kv<<<(2048 * 1024) / 256, 256, 0, stream>>>(qkv, k_out, v_out);

  // 3) flash attention: 128 query blocks x (16 heads = 16 waves)
  attn_head<<<dim3(128), 512, 0, stream>>>(qkv, mask, shift, att);

  // 4) o = att @ Wo + bo  (M=2048, N=1024, K=1024)
  gemm_bf16_wmma<<<dim3(1024 / BN, 2048 / BM), 256, 0, stream>>>(
      att, Wo, bo, out_o, 2048, 1024, 1024);
}
